// Quantizer2D_39402029974035
// MI455X (gfx1250) — compile-verified, matrix operand order checked by reading, not run on hardware
//
#include <hip/hip_runtime.h>

// ---------------------------------------------------------------------------
// CDNA5 (gfx1250) fused VQ-VAE encoder + vector-quantizer.
// wave32, WMMA bf16 16x16x32; 32 rows per wave (2 A-tiles share each B-tile);
// W2^T and codebook chunks staged in LDS; async-to-LDS + prefetch paths.
// ---------------------------------------------------------------------------

typedef __attribute__((ext_vector_type(16))) __bf16 v16bf;
typedef __attribute__((ext_vector_type(8)))  __bf16 v8bf;
typedef __attribute__((ext_vector_type(8)))  float  v8f;
typedef int v4i_vs __attribute__((vector_size(16)));   // gcc-style, matches builtin

#ifndef __has_builtin
#define __has_builtin(x) 0
#endif
#if __has_builtin(__builtin_amdgcn_global_load_async_to_lds_b128) && \
    __has_builtin(__builtin_amdgcn_s_wait_asynccnt)
#define HAVE_ASYNC_LDS 1
#else
#define HAVE_ASYNC_LDS 0
#endif

#define HDIM   256
#define DDIM   128
#define KCODES 1024
#define MBLK   256      // rows per workgroup (8 waves x 32 rows)
#define HPAD   264      // 256 + 8 halfs pad -> conflict-free ds_load_b128
#define ZPAD   136      // 128 + 8 halfs pad
#define CCHUNK 256      // codebook rows resident in LDS per pass

// LDS carve (bytes)
#define OFF_HBUF   0
#define SZ_HBUF    (MBLK * HPAD * 2)            // 135168
#define OFF_ZBUF   (OFF_HBUF + SZ_HBUF)
#define SZ_ZBUF    (MBLK * ZPAD * 2)            // 69632
#define OFF_CBUF   (OFF_ZBUF + SZ_ZBUF)
#define SZ_CBUF    (CCHUNK * ZPAD * 2)          // 69632 (aliased: W2^T in ph.2)
#define OFF_CNORM  (OFF_CBUF + SZ_CBUF)
#define SZ_CNORM   (KCODES * 4)                 // 4096
#define OFF_RIDX   (OFF_CNORM + SZ_CNORM)
#define SZ_RIDX    (MBLK * 4)                   // 1024
#define OFF_RSCR   (OFF_RIDX + SZ_RIDX)
#define SZ_RSCR    (MBLK * 4)                   // 1024
#define SMEM_BYTES (OFF_RSCR + SZ_RSCR)         // 280576 B  (< 320KB / WGP)

#if HAVE_ASYNC_LDS
// prototype (from toolchain diagnostic): (int4 AS1* gsrc, int4 AS3* lds, imm, imm)
#define ASYNC_CP16(dst, src)                                                   \
  __builtin_amdgcn_global_load_async_to_lds_b128(                              \
      (__attribute__((address_space(1))) v4i_vs*)(src),                        \
      (__attribute__((address_space(3))) v4i_vs*)(dst), 0, 0)
#endif

// 16x32 bf16 WMMA operand tile (A layout; B uses the mirrored N<->lane
// mapping which is the identical address pattern).
// lanes 0-15 : row = row0+lane,    K halves {0..7, 16..23}
// lanes 16-31: row = row0+lane-16, K halves {8..15, 24..31}
__device__ __forceinline__ v16bf load_tile16(const __bf16* base, int row0,
                                             int rowStride, int kbase, int lane) {
  const int half = lane >> 4;
  const int r    = row0 + (lane & 15);
  const __bf16* p = base + r * rowStride + kbase + half * 8;
  v8bf lo = *(const v8bf*)(p);
  v8bf hi = *(const v8bf*)(p + 16);
  return __builtin_shufflevector(lo, hi, 0,1,2,3,4,5,6,7,8,9,10,11,12,13,14,15);
}

__device__ __forceinline__ v8f wmma_bf16(v16bf a, v16bf b, v8f c) {
  return __builtin_amdgcn_wmma_f32_16x16x32_bf16(false, a, false, b,
                                                 (short)0, c, false, false);
}

// ---------------------------------------------------------------------------
// Prep kernels: bf16 W2^T, bf16 codebook copy + per-code norms, zero loss.
// ---------------------------------------------------------------------------
__global__ __launch_bounds__(256) void prep_w2t(const float* __restrict__ W2,
                                                __bf16* __restrict__ w2t,
                                                float* __restrict__ out_loss) {
  int e = blockIdx.x * 256 + threadIdx.x;      // 256*128 elements
  int h = e >> 7;                              // W2 is [H=256][D=128] row-major
  int d = e & 127;
  w2t[d * HDIM + h] = (__bf16)W2[e];
  if (e == 0) *out_loss = 0.0f;
}

__global__ __launch_bounds__(128) void prep_cb(const float* __restrict__ cb,
                                               __bf16* __restrict__ cb16,
                                               float* __restrict__ norms) {
  __shared__ float red[128];
  int code = blockIdx.x, t = threadIdx.x;
  float v = cb[code * DDIM + t];
  cb16[code * DDIM + t] = (__bf16)v;
  red[t] = v * v;
  __syncthreads();
  for (int s = 64; s > 0; s >>= 1) {
    if (t < s) red[t] += red[t + s];
    __syncthreads();
  }
  if (t == 0) norms[code] = red[0];
}

// ---------------------------------------------------------------------------
// Fused main kernel. Block = 256 threads = 8 waves; 256 rows per block.
// ---------------------------------------------------------------------------
__global__ __launch_bounds__(256) void vq_fused(
    const int*    __restrict__ xy,
    const float*  __restrict__ W1,      // [2][256]
    const float*  __restrict__ b1,
    const float*  __restrict__ ln_g,
    const float*  __restrict__ ln_b,
    const float*  __restrict__ b2,      // [128]
    const float*  __restrict__ cb_f32,  // [1024][128]
    const __bf16* __restrict__ w2t,     // [128][256] bf16 (W2 transposed)
    const __bf16* __restrict__ cb16,    // [1024][128] bf16
    const float*  __restrict__ cnorm_g, // [1024]
    float* __restrict__ out_q,
    float* __restrict__ out_idx,
    float* __restrict__ out_loss,
    int N)
{
  extern __shared__ char smem[];
  __bf16* hbuf   = (__bf16*)(smem + OFF_HBUF);
  __bf16* zbuf   = (__bf16*)(smem + OFF_ZBUF);
  __bf16* cbuf   = (__bf16*)(smem + OFF_CBUF);   // W2^T during ph.2, codebook ph.3
  float*  cnormS = (float*) (smem + OFF_CNORM);
  int*    rowidx = (int*)   (smem + OFF_RIDX);
  float*  rscr   = (float*) (smem + OFF_RSCR);

  const int tid  = threadIdx.x;
  const int lane = tid & 31;
  const int wave = tid >> 5;
  const int lh   = lane >> 4;
  const int m0   = blockIdx.x * MBLK;
  const int r0   = wave * 32;                 // 32 rows per wave

  // stage code norms (stays in LDS for the whole kernel)
  for (int i = tid; i < KCODES; i += 256) cnormS[i] = cnorm_g[i];

  // stage W2^T into the (currently unused) cbuf region: 128 rows x 256 halfs,
  // padded to HPAD for conflict-free B loads in phase 2.
  {
#if HAVE_ASYNC_LDS
    for (int i = tid; i < (DDIM * HDIM) / 8; i += 256) {   // 4096 v8bf
      int row = i >> 5, c8 = i & 31;
      ASYNC_CP16(cbuf + row * HPAD + c8 * 8, w2t + row * HDIM + c8 * 8);
    }
    __builtin_amdgcn_s_wait_asynccnt(0);
#else
    for (int i = tid; i < (DDIM * HDIM) / 8; i += 256) {
      int row = i >> 5, c8 = i & 31;
      *(v8bf*)(cbuf + row * HPAD + c8 * 8) =
          *(const v8bf*)(w2t + row * HDIM + c8 * 8);
    }
#endif
  }

  // ---- Phase 1: encoder layer 1 + LayerNorm + ReLU -> bf16 h tile in LDS
  {
    const int m = m0 + tid;
    const float a = (float)xy[2 * m]     * (2.0f / 639.0f) - 1.0f;
    const float b = (float)xy[2 * m + 1] * (2.0f / 479.0f) - 1.0f;
    float s1 = 0.0f, s2 = 0.0f;
    for (int j = 0; j < HDIM; ++j) {
      float h = fmaf(a, W1[j], fmaf(b, W1[HDIM + j], b1[j]));
      s1 += h; s2 += h * h;
    }
    const float mu  = s1 * (1.0f / HDIM);
    const float var = s2 * (1.0f / HDIM) - mu * mu;
    const float rs  = rsqrtf(var + 1e-5f);
    __bf16* hrow = hbuf + tid * HPAD;
    for (int j = 0; j < HDIM; ++j) {
      float h = fmaf(a, W1[j], fmaf(b, W1[HDIM + j], b1[j]));
      float v = (h - mu) * rs * ln_g[j] + ln_b[j];
      hrow[j] = (__bf16)(v > 0.0f ? v : 0.0f);
    }
  }
  __syncthreads();

  // ---- Phase 2: z = relu_h @ W2 + b2 ; two 16x128 tiles per wave,
  //      every B tile (from LDS W2^T) feeds two WMMAs.
  v8f acc0[8], acc1[8];
#pragma unroll
  for (int nt = 0; nt < 8; ++nt) { acc0[nt] = (v8f){}; acc1[nt] = (v8f){}; }
  for (int kk = 0; kk < 8; ++kk) {
    v16bf A0 = load_tile16(hbuf, r0,      HPAD, kk * 32, lane);
    v16bf A1 = load_tile16(hbuf, r0 + 16, HPAD, kk * 32, lane);
#pragma unroll
    for (int nt = 0; nt < 8; ++nt) {
      v16bf B = load_tile16(cbuf, nt * 16, HPAD, kk * 32, lane);
      acc0[nt] = wmma_bf16(A0, B, acc0[nt]);
      acc1[nt] = wmma_bf16(A1, B, acc1[nt]);
    }
  }
  // add bias, spill z (bf16) row-major [row][d] for phase-3 A loads
#pragma unroll
  for (int nt = 0; nt < 8; ++nt) {
    const float bb = b2[nt * 16 + (lane & 15)];
#pragma unroll
    for (int v = 0; v < 8; ++v) {
      zbuf[(r0 + v + 8 * lh) * ZPAD + nt * 16 + (lane & 15)] =
          (__bf16)(acc0[nt][v] + bb);
      zbuf[(r0 + 16 + v + 8 * lh) * ZPAD + nt * 16 + (lane & 15)] =
          (__bf16)(acc1[nt][v] + bb);
    }
  }
  __syncthreads();

  // ---- Phase 3: distances to 1024 codes, codebook streamed via LDS chunks
  v16bf Az0[4], Az1[4];
#pragma unroll
  for (int kk = 0; kk < 4; ++kk) {
    Az0[kk] = load_tile16(zbuf, r0,      ZPAD, kk * 32, lane);
    Az1[kk] = load_tile16(zbuf, r0 + 16, ZPAD, kk * 32, lane);
  }

  float mv0[8], mv1[8]; int mi0[8], mi1[8];
#pragma unroll
  for (int v = 0; v < 8; ++v) {
    mv0[v] = 3.4e38f; mi0[v] = 0; mv1[v] = 3.4e38f; mi1[v] = 0;
  }

  for (int cc = 0; cc < KCODES / CCHUNK; ++cc) {
    __syncthreads();                     // previous chunk fully consumed
    const __bf16* src = cb16 + (size_t)cc * CCHUNK * DDIM;
#if HAVE_ASYNC_LDS
    for (int i = tid; i < CCHUNK * DDIM / 8; i += 256) {
      int row = i >> 4, c8 = i & 15;
      ASYNC_CP16(cbuf + row * ZPAD + c8 * 8, src + row * DDIM + c8 * 8);
    }
    __builtin_amdgcn_s_wait_asynccnt(0);
#else
    for (int i = tid; i < CCHUNK * DDIM / 8; i += 256) {
      int row = i >> 4, c8 = i & 15;
      *(v8bf*)(cbuf + row * ZPAD + c8 * 8) =
          *(const v8bf*)(src + row * DDIM + c8 * 8);
    }
#endif
    // prefetch next chunk toward L2/L0 while this one is consumed
    if (cc + 1 < KCODES / CCHUNK)
      __builtin_prefetch(cb16 + (size_t)(cc + 1) * CCHUNK * DDIM + tid * 128, 0, 1);
    __syncthreads();

    for (int ct = 0; ct < CCHUNK / 16; ++ct) {
      v8f s0 = (v8f){}, s1 = (v8f){};
#pragma unroll
      for (int kk = 0; kk < 4; ++kk) {
        v16bf B = load_tile16(cbuf, ct * 16, ZPAD, kk * 32, lane);
        s0 = wmma_bf16(Az0[kk], B, s0);
        s1 = wmma_bf16(Az1[kk], B, s1);
      }
      const int gcb = cc * CCHUNK + ct * 16 + (lane & 15);
      const float nc = cnormS[gcb];
#pragma unroll
      for (int v = 0; v < 8; ++v) {
        float d0 = fmaf(-2.0f, s0[v], nc);   // ||z||^2 row-constant: dropped
        float d1 = fmaf(-2.0f, s1[v], nc);
        if (d0 < mv0[v]) { mv0[v] = d0; mi0[v] = gcb; }
        if (d1 < mv1[v]) { mv1[v] = d1; mi1[v] = gcb; }
      }
    }
  }

  // cross-lane argmin within each 16-lane group (wave32 shuffles)
#pragma unroll
  for (int v = 0; v < 8; ++v) {
    float bv = mv0[v]; int bi = mi0[v];
    for (int off = 8; off > 0; off >>= 1) {
      float ov = __shfl_xor(bv, off, 16);
      int   oi = __shfl_xor(bi, off, 16);
      if (ov < bv || (ov == bv && oi < bi)) { bv = ov; bi = oi; }
    }
    if ((lane & 15) == 0) rowidx[r0 + v + 8 * lh] = bi;

    bv = mv1[v]; bi = mi1[v];
    for (int off = 8; off > 0; off >>= 1) {
      float ov = __shfl_xor(bv, off, 16);
      int   oi = __shfl_xor(bi, off, 16);
      if (ov < bv || (ov == bv && oi < bi)) { bv = ov; bi = oi; }
    }
    if ((lane & 15) == 0) rowidx[r0 + 16 + v + 8 * lh] = bi;
  }
  __syncthreads();

  // ---- Phase 4: gather q rows (f32 codebook, L2-resident), idx, loss
  {
    const int r = tid, m = m0 + r;
    const int k = rowidx[r];
    const float*  crow = cb_f32 + (size_t)k * DDIM;
    const __bf16* zrow = zbuf + r * ZPAD;
    float* qrow = out_q + (size_t)m * DDIM;
    float lsum = 0.0f;
    for (int d = 0; d < DDIM; d += 4) {
      float4 q4 = *(const float4*)(crow + d);
      float e0 = q4.x - (float)zrow[d + 0];
      float e1 = q4.y - (float)zrow[d + 1];
      float e2 = q4.z - (float)zrow[d + 2];
      float e3 = q4.w - (float)zrow[d + 3];
      lsum += e0 * e0 + e1 * e1 + e2 * e2 + e3 * e3;
      *(float4*)(qrow + d) = q4;
    }
    out_idx[m] = (float)k;
    rscr[r] = lsum;
  }
  __syncthreads();
  if (tid == 0) {
    float t = 0.0f;
    for (int i = 0; i < MBLK; ++i) t += rscr[i];
    atomicAdd(out_loss, 1.25f * t / ((float)N * (float)DDIM));
  }
}

// ---------------------------------------------------------------------------
extern "C" void kernel_launch(void* const* d_in, const int* in_sizes, int n_in,
                              void* d_out, int out_size, void* d_ws, size_t ws_size,
                              hipStream_t stream) {
  const int*   xy   = (const int*)  d_in[0];
  const float* W1   = (const float*)d_in[1];
  const float* b1   = (const float*)d_in[2];
  const float* ln_g = (const float*)d_in[3];
  const float* ln_b = (const float*)d_in[4];
  const float* W2   = (const float*)d_in[5];
  const float* b2   = (const float*)d_in[6];
  const float* cb   = (const float*)d_in[7];

  const int N = in_sizes[0] / 2;                       // 262144

  float* out      = (float*)d_out;
  float* out_q    = out;                               // [N][128]
  float* out_idx  = out + (size_t)N * DDIM;            // [N]
  float* out_loss = out + (size_t)N * DDIM + N;        // scalar

  __bf16* w2t   = (__bf16*)d_ws;                                       // 64KB
  __bf16* cb16  = (__bf16*)((char*)d_ws + 65536);                      // 256KB
  float*  norms = (float*) ((char*)d_ws + 65536 + 262144);             // 4KB

  prep_w2t<<<(HDIM * DDIM) / 256, 256, 0, stream>>>(W2, w2t, out_loss);
  prep_cb<<<KCODES, 128, 0, stream>>>(cb, cb16, norms);
  vq_fused<<<N / MBLK, 256, SMEM_BYTES, stream>>>(
      xy, W1, b1, ln_g, ln_b, b2, cb, w2t, cb16, norms,
      out_q, out_idx, out_loss, N);
}
